// MMOE_IN_DIEN_59528246722874
// MI455X (gfx1250) — matrix operand order repaired and checked by reading, not compiled, not run on hardware
//
#include <hip/hip_runtime.h>
#include <hip/hip_bf16.h>
#include <math.h>

typedef __attribute__((ext_vector_type(16))) _Float16 v16h;
typedef __attribute__((ext_vector_type(8)))  _Float16 v8h;
typedef __attribute__((ext_vector_type(8)))  float    v8f;
typedef __attribute__((ext_vector_type(4)))  uint32_t u32x4;
typedef __attribute__((ext_vector_type(8)))  int      i32x8;
typedef __attribute__((ext_vector_type(4)))  int      i32x4;

#define B_SZ   16384
#define T_SZ   50
#define EMB    16
#define NS     8
#define ND     8
#define NEXP   4
#define DNN_DIM 184
#define K1PAD  192
#define NEG_INF_F (-4294967295.0f)

__device__ __forceinline__ v8f wmma16(const v16h& a, const v16h& b, const v8f& c) {
  return __builtin_amdgcn_wmma_f32_16x16x32_f16(false, a, false, b, (short)0, c, false, false);
}
__device__ __forceinline__ v16h cat8(const v8h& lo, const v8h& hi) {
  return __builtin_shufflevector(lo, hi, 0, 1, 2, 3, 4, 5, 6, 7,
                                         8, 9, 10, 11, 12, 13, 14, 15);
}

// 1-D contiguous TDM copy global -> LDS, data_size = 8 bytes.
// D# group0: count=1 | lds_addr | global_addr[56:0] | type=2
// D# group1: data_size=3(8B); tensor_dim0 = tile_dim0 = n elems; dim1 = 1
__device__ __forceinline__ void tdm_copy_1d(const void* gsrc, void* lds_dst,
                                            uint32_t nbytes) {
  uint32_t n = nbytes >> 3;
  uint64_t ga = (uint64_t)(uintptr_t)gsrc;
  uint32_t la = (uint32_t)(uintptr_t)lds_dst;   // low 32 bits = LDS offset
  u32x4 g0;
  g0[0] = 1u;                                          // count=1 (valid user D#)
  g0[1] = la;                                          // lds_addr
  g0[2] = (uint32_t)(ga & 0xFFFFFFFFu);                // global_addr[31:0]
  g0[3] = (uint32_t)((ga >> 32) & 0x1FFFFFFu) | (2u << 30);  // [56:32] | type=2
  i32x8 g1;
  g1[0] = (int)(3u << 16);                             // wg_mask=0, data_size=8B
  g1[1] = (int)((n & 0xFFFFu) << 16);                  // tensor_dim0[15:0]
  g1[2] = (int)((n >> 16) & 0xFFFFu) | (int)(1u << 16);// tensor_dim0[31:16], dim1=1
  g1[3] = (int)((n & 0xFFFFu) << 16);                  // tile_dim0 = n
  g1[4] = 1;                                           // tile_dim1 = 1
  g1[5] = (int)n;                                      // tensor_dim0_stride[31:0]
  g1[6] = 0;
  g1[7] = 0;
  i32x4 z4 = {0, 0, 0, 0};
  i32x8 z8 = {0, 0, 0, 0, 0, 0, 0, 0};
  __builtin_amdgcn_tensor_load_to_lds(g0, g1, z4, z4, z8, 0);
}

// ---------------------------------------------------------------------------
// Kernel 0: one-time weight convert/transpose to f16 tile-contiguous blobs
//   ew1T: [e][nt(8)][16 n][192 k]   ew2T: [e][nt(4)][16 n][128 k]
//   dinW: w1T [64 n][64 k] then w2T [32 n][64 k]
// ---------------------------------------------------------------------------
#define N_EW1T (NEXP * 8 * 16 * K1PAD)   /* 98304 */
#define N_EW2T (NEXP * 4 * 16 * 128)     /* 32768  */
#define N_DINW (64 * 64 + 32 * 64)       /* 6144   */

__global__ __launch_bounds__(256) void k_prep(
    const float* __restrict__ ew1, const float* __restrict__ ew2,
    const float* __restrict__ w1, const float* __restrict__ w2,
    _Float16* __restrict__ ew1T, _Float16* __restrict__ ew2T,
    _Float16* __restrict__ dinW)
{
  int i = blockIdx.x * 256 + threadIdx.x;
  if (i < N_EW1T) {
    int k = i % K1PAD;
    int t = i / K1PAD;                 // (e*8+nt)*16 + n
    int n = t & 15, tile = t >> 4;
    int nt = tile & 7, e = tile >> 3;
    float v = (k < DNN_DIM) ? ew1[((size_t)e * DNN_DIM + k) * 128 + nt * 16 + n] : 0.f;
    ew1T[i] = (_Float16)v;
  } else if (i < N_EW1T + N_EW2T) {
    int j = i - N_EW1T;
    int k = j & 127;
    int t = j >> 7;
    int n = t & 15, tile = t >> 4;
    int nt = tile & 3, e = tile >> 2;
    ew2T[j] = (_Float16)ew2[((size_t)e * 128 + k) * 64 + nt * 16 + n];
  } else if (i < N_EW1T + N_EW2T + N_DINW) {
    int j = i - N_EW1T - N_EW2T;
    if (j < 4096) {
      int n = j >> 6, k = j & 63;
      dinW[j] = (_Float16)w1[k * 64 + n];
    } else {
      int jj = j - 4096;
      int n = jj >> 6, k = jj & 63;
      dinW[j] = (_Float16)w2[k * 32 + n];
    }
  }
}

// ---------------------------------------------------------------------------
// Kernel 1: dense copy + sparse gathers + seq mean-pool -> dnn[:, 0:152)
// ---------------------------------------------------------------------------
__global__ __launch_bounds__(256) void k_embed(
    const float* __restrict__ dense, const float* __restrict__ tables,
    const float* __restrict__ seq_table, const int* __restrict__ sparse_idx,
    const int* __restrict__ seq_idx, float* __restrict__ dnn, int V)
{
  int wv   = threadIdx.x >> 5;
  int lane = threadIdx.x & 31;
  int b    = blockIdx.x * 8 + wv;
  int j = lane & 15, p = lane >> 4;
  float* row = dnn + (size_t)b * DNN_DIM;

  if (lane < ND) row[lane] = dense[b * ND + lane];

  for (int s = p; s < NS; s += 2) {
    int idx = sparse_idx[b * NS + s];
    row[ND + s * EMB + j] = tables[((size_t)s * V + idx) * EMB + j];
  }

  float acc = 0.f;
  for (int t = p; t < T_SZ; t += 2) {
    int idx = seq_idx[b * T_SZ + t];
    acc += seq_table[(size_t)idx * EMB + j];
  }
  acc += __shfl_xor(acc, 16);
  if (p == 0) row[ND + NS * EMB + j] = acc * (1.0f / T_SZ);
}

// ---------------------------------------------------------------------------
// Kernel 2: DIN attention (WMMA f16 MLP + softmax + weighted sum) -> [152:168)
// Weights arrive via TDM (overlapped with gathers/feature build).
// ---------------------------------------------------------------------------
__global__ __launch_bounds__(128) void k_din(
    const float* __restrict__ table0, const int* __restrict__ sparse_idx,
    const int* __restrict__ hist, const _Float16* __restrict__ dinW,
    const float* __restrict__ b1, const float* __restrict__ b2,
    const float* __restrict__ w3, const float* __restrict__ b3,
    float* __restrict__ dnn)
{
  __shared__ float    qs[EMB];
  __shared__ float    klds[64][EMB];
  __shared__ float    maskv[64];
  __shared__ __align__(32) _Float16 wblob[6144];      // w1T (4096) + w2T (2048)
  __shared__ __align__(32) _Float16 featl[64 * 64];   // [t][k]   8KB
  __shared__ __align__(32) _Float16 h1l[4][16 * 64];  // [t][k]   8KB
  __shared__ float    w3s[32], b1s[64], b2s[32];
  __shared__ float    scw[4][16];
  __shared__ float    scores[64];
  __shared__ float    wexp[64];

  int tid = threadIdx.x;
  int b   = blockIdx.x;

  // async: pull pre-transposed f16 weights into LDS via Tensor Data Mover
  if (tid < 32) tdm_copy_1d(dinW, wblob, 6144 * 2);

  if (tid < 32) { w3s[tid] = w3[tid]; b2s[tid] = b2[tid]; }
  if (tid < 64) { b1s[tid] = b1[tid]; scw[tid >> 4][tid & 15] = 0.f; }
  if (tid < EMB) {
    int tg = sparse_idx[b * NS + 0];
    qs[tid] = table0[(size_t)tg * EMB + tid];
  }
  for (int i = tid; i < 64 * EMB; i += 128) {
    int t = i >> 4, jj = i & 15;
    float v = 0.f;
    if (t < T_SZ) {
      int hi = hist[b * T_SZ + t];
      v = table0[(size_t)hi * EMB + jj];
    }
    klds[t][jj] = v;
  }
  if (tid < 64) {
    float m = 0.f;
    if (tid < T_SZ) m = (hist[b * T_SZ + tid] != 0) ? 1.f : 0.f;
    maskv[tid] = m;
  }
  __syncthreads();

  // attention features row-major: featl[t][k], k in [q, k, q-k, q*k]
  for (int i = tid; i < 64 * 64; i += 128) {
    int t = i >> 6, k = i & 63;
    float f;
    if      (k < 16) f = qs[k];
    else if (k < 32) f = klds[t][k - 16];
    else if (k < 48) f = qs[k - 32] - klds[t][k - 32];
    else             f = qs[k - 48] * klds[t][k - 48];
    featl[i] = (_Float16)f;
  }
  if (tid < 32) __builtin_amdgcn_s_wait_tensorcnt(0);
  __syncthreads();

  const _Float16* w1T = wblob;          // [n][k] 64x64
  const _Float16* w2T = wblob + 4096;   // [n][k] 32x64

  int wv = tid >> 5, lane = tid & 31;
  int l15 = lane & 15, hw = lane >> 4;
  int t_row = wv * 16 + l15;

  // Layer 1 (transposed): D^T[n][t]
  for (int nt = 0; nt < 4; ++nt) {
    int n = nt * 16 + l15;
    v8f acc = {};
    #pragma unroll
    for (int kc = 0; kc < 2; ++kc) {
      v8h alo = *(const v8h*)&w1T[n * 64 + kc * 32 + 8 * hw];
      v8h ahi = *(const v8h*)&w1T[n * 64 + kc * 32 + 16 + 8 * hw];
      v16h bv = cat8(*(const v8h*)&featl[t_row * 64 + kc * 32 + 8 * hw], *(const v8h*)&featl[t_row * 64 + kc * 32 + 16 + 8 * hw]);
      acc = wmma16(cat8(alo, ahi), bv, acc);
    }
    v8h hv;
    #pragma unroll
    for (int r = 0; r < 8; ++r)
      hv[r] = (_Float16)fmaxf(acc[r] + b1s[nt * 16 + r + 8 * hw], 0.f);
    *(v8h*)&h1l[wv][l15 * 64 + nt * 16 + 8 * hw] = hv;
  }

  // Layer 2 (transposed) + fused w3 dot -> one ds_add_f32 per lane/tile
  for (int nt = 0; nt < 2; ++nt) {
    int n = nt * 16 + l15;
    v8f acc = {};
    #pragma unroll
    for (int kc = 0; kc < 2; ++kc) {
      v8h alo = *(const v8h*)&w2T[n * 64 + kc * 32 + 8 * hw];
      v8h ahi = *(const v8h*)&w2T[n * 64 + kc * 32 + 16 + 8 * hw];
      v16h bv = cat8(*(const v8h*)&h1l[wv][l15 * 64 + kc * 32 + 8 * hw], *(const v8h*)&h1l[wv][l15 * 64 + kc * 32 + 16 + 8 * hw]);
      acc = wmma16(cat8(alo, ahi), bv, acc);
    }
    float part = 0.f;
    #pragma unroll
    for (int r = 0; r < 8; ++r) {
      int nn = nt * 16 + r + 8 * hw;
      part += fmaxf(acc[r] + b2s[nn], 0.f) * w3s[nn];
    }
    atomicAdd(&scw[wv][l15], part);
  }

  if (lane < 16) {
    float s = scw[wv][l15] + b3[0];
    float sc;
    if (t_row >= T_SZ) sc = -INFINITY;
    else               sc = (maskv[t_row] > 0.f) ? s : NEG_INF_F;
    scores[t_row] = sc;
  }
  __syncthreads();

  if (wv == 0) {
    float s1 = scores[lane], s2 = scores[lane + 32];
    float mx = fmaxf(s1, s2);
    #pragma unroll
    for (int o = 16; o >= 1; o >>= 1) mx = fmaxf(mx, __shfl_xor(mx, o));
    float e1 = __expf(s1 - mx);
    float e2 = (lane + 32 >= T_SZ) ? 0.f : __expf(s2 - mx);
    wexp[lane] = e1; wexp[lane + 32] = e2;
    float ssum = e1 + e2;
    #pragma unroll
    for (int o = 16; o >= 1; o >>= 1) ssum += __shfl_xor(ssum, o);
    float inv = 1.0f / ssum;
    if (lane < EMB) {
      float acc2 = 0.f;
      for (int t = 0; t < T_SZ; ++t) acc2 += wexp[t] * klds[t][lane];
      dnn[(size_t)b * DNN_DIM + ND + NS * EMB + EMB + lane] = acc2 * inv;
    }
  }
}

// ---------------------------------------------------------------------------
// Kernel 3: GRU (reset_after), hidden=16, sequential over T -> [168:184)
// ---------------------------------------------------------------------------
__global__ __launch_bounds__(256) void k_gru(
    const float* __restrict__ table0, const int* __restrict__ hist,
    const float* __restrict__ Wg, const float* __restrict__ Ug,
    const float* __restrict__ bg, float* __restrict__ dnn)
{
  __shared__ float Ws[16 * 48], Us[16 * 48], bis[48], brs[48];
  __shared__ float kbuf[16][16], hbuf[16][16];
  int tid = threadIdx.x;
  for (int i = tid; i < 16 * 48; i += 256) { Ws[i] = Wg[i]; Us[i] = Ug[i]; }
  if (tid < 48) { bis[tid] = bg[tid]; brs[tid] = bg[48 + tid]; }
  int r = tid >> 4, j = tid & 15;
  int b = blockIdx.x * 16 + r;
  hbuf[r][j] = 0.f;
  float h = 0.f;
  __syncthreads();

  for (int t = 0; t < T_SZ; ++t) {
    int hi = hist[b * T_SZ + t];
    kbuf[r][j] = table0[(size_t)hi * EMB + j];
    __syncthreads();
    float xz = bis[j], xr = bis[16 + j], xh = bis[32 + j];
    float rz = brs[j], rr = brs[16 + j], rh = brs[32 + j];
    #pragma unroll
    for (int i = 0; i < 16; ++i) {
      float kv = kbuf[r][i], hv = hbuf[r][i];
      xz += kv * Ws[i * 48 + j];
      xr += kv * Ws[i * 48 + 16 + j];
      xh += kv * Ws[i * 48 + 32 + j];
      rz += hv * Us[i * 48 + j];
      rr += hv * Us[i * 48 + 16 + j];
      rh += hv * Us[i * 48 + 32 + j];
    }
    float z  = 1.f / (1.f + __expf(-(xz + rz)));
    float rg = 1.f / (1.f + __expf(-(xr + rr)));
    float hh = tanhf(xh + rg * rh);
    float hn = z * h + (1.f - z) * hh;
    if (hi != 0) h = hn;
    __syncthreads();
    hbuf[r][j] = h;
  }
  dnn[(size_t)b * DNN_DIM + ND + NS * EMB + 2 * EMB + j] = h;
}

// ---------------------------------------------------------------------------
// Kernel 4: MMOE experts (WMMA f16) + gates + heads -> d_out
// Weight tiles arrive via TDM from the tile-contiguous f16 blobs.
// ---------------------------------------------------------------------------
__global__ __launch_bounds__(128) void k_mmoe(
    const float* __restrict__ dnn,
    const _Float16* __restrict__ ew1T, const _Float16* __restrict__ ew2T,
    const float* __restrict__ eb1, const float* __restrict__ eb2,
    const float* __restrict__ gfw, const float* __restrict__ gfb,
    const float* __restrict__ glw, const float* __restrict__ glb,
    const float* __restrict__ fw,  const float* __restrict__ fb,
    const float* __restrict__ lw,  const float* __restrict__ lb,
    float* __restrict__ out)
{
  __shared__ __align__(32) _Float16 xs[64 * K1PAD];   // [row][k]  24KB
  __shared__ __align__(32) _Float16 h1l[64 * 128];    // [row][k]  16KB
  __shared__ __align__(32) _Float16 w1t[16 * K1PAD];  // [n][k]     6KB
  __shared__ __align__(32) _Float16 w2t[16 * 128];    // [n][k]     4KB
  __shared__ float eb1s[128], eb2s[64], fws[64], lws[64];
  __shared__ float dotf[4][16], dotl[4][16];

  int tid = threadIdx.x;
  int rowblk = blockIdx.x * 64;
  for (int i = tid; i < 64 * K1PAD; i += 128) {
    int r = i / K1PAD, k = i - r * K1PAD;
    float v = (k < DNN_DIM) ? dnn[(size_t)(rowblk + r) * DNN_DIM + k] : 0.f;
    xs[i] = (_Float16)v;
  }
  if (tid < 64) { fws[tid] = fw[tid]; lws[tid] = lw[tid]; }

  int wv = tid >> 5, lane = tid & 31;
  int l15 = lane & 15, hw = lane >> 4;
  if (lane < 16) { dotf[wv][l15] = 0.f; dotl[wv][l15] = 0.f; }

  // gates (per-row 184-dot for 4 experts x 2 heads), split across lane halves
  int grow = rowblk + wv * 16 + l15;
  float pf[4] = {0, 0, 0, 0}, pl[4] = {0, 0, 0, 0};
  for (int f = hw * 92; f < hw * 92 + 92; ++f) {
    float xv = dnn[(size_t)grow * DNN_DIM + f];
    #pragma unroll
    for (int e = 0; e < 4; ++e) {
      pf[e] += xv * gfw[f * 4 + e];
      pl[e] += xv * glw[f * 4 + e];
    }
  }
  float gF[4], gL[4];
  #pragma unroll
  for (int e = 0; e < 4; ++e) {
    gF[e] = pf[e] + __shfl_xor(pf[e], 16) + gfb[e];
    gL[e] = pl[e] + __shfl_xor(pl[e], 16) + glb[e];
  }
  {
    float mxf = fmaxf(fmaxf(gF[0], gF[1]), fmaxf(gF[2], gF[3]));
    float mxl = fmaxf(fmaxf(gL[0], gL[1]), fmaxf(gL[2], gL[3]));
    float sf = 0.f, sl = 0.f;
    #pragma unroll
    for (int e = 0; e < 4; ++e) {
      gF[e] = __expf(gF[e] - mxf); sf += gF[e];
      gL[e] = __expf(gL[e] - mxl); sl += gL[e];
    }
    float invf = 1.f / sf, invl = 1.f / sl;
    #pragma unroll
    for (int e = 0; e < 4; ++e) { gF[e] *= invf; gL[e] *= invl; }
  }

  float accF = 0.f, accL = 0.f;
  for (int e = 0; e < NEXP; ++e) {
    __syncthreads();
    if (tid < 128) eb1s[tid] = eb1[e * 128 + tid];
    if (tid < 64)  eb2s[tid] = eb2[e * 64 + tid];

    // Layer 1: h1^T = w1^T x X^T, per wave 16-row tile
    for (int nt = 0; nt < 8; ++nt) {
      __syncthreads();
      if (wv == 0) {
        tdm_copy_1d(ew1T + (size_t)(e * 8 + nt) * 16 * K1PAD, w1t, 16 * K1PAD * 2);
        __builtin_amdgcn_s_wait_tensorcnt(0);
      }
      __syncthreads();
      v8f acc = {};
      #pragma unroll
      for (int kc = 0; kc < 6; ++kc) {
        v8h alo = *(const v8h*)&w1t[l15 * K1PAD + kc * 32 + 8 * hw];
        v8h ahi = *(const v8h*)&w1t[l15 * K1PAD + kc * 32 + 16 + 8 * hw];
        v16h bv = cat8(*(const v8h*)&xs[(wv * 16 + l15) * K1PAD + kc * 32 + 8 * hw], *(const v8h*)&xs[(wv * 16 + l15) * K1PAD + kc * 32 + 16 + 8 * hw]);
        acc = wmma16(cat8(alo, ahi), bv, acc);
      }
      v8h hv;
      #pragma unroll
      for (int r = 0; r < 8; ++r)
        hv[r] = (_Float16)fmaxf(acc[r] + eb1s[nt * 16 + r + 8 * hw], 0.f);
      *(v8h*)&h1l[(wv * 16 + l15) * 128 + nt * 16 + 8 * hw] = hv;
    }

    // Layer 2 + fused head dots (one ds_add_f32 per head per lane/tile)
    for (int nt = 0; nt < 4; ++nt) {
      __syncthreads();
      if (wv == 0) {
        tdm_copy_1d(ew2T + (size_t)(e * 4 + nt) * 16 * 128, w2t, 16 * 128 * 2);
        __builtin_amdgcn_s_wait_tensorcnt(0);
      }
      __syncthreads();
      v8f acc = {};
      #pragma unroll
      for (int kc = 0; kc < 4; ++kc) {
        v8h alo = *(const v8h*)&w2t[l15 * 128 + kc * 32 + 8 * hw];
        v8h ahi = *(const v8h*)&w2t[l15 * 128 + kc * 32 + 16 + 8 * hw];
        v16h bv = cat8(*(const v8h*)&h1l[(wv * 16 + l15) * 128 + kc * 32 + 8 * hw], *(const v8h*)&h1l[(wv * 16 + l15) * 128 + kc * 32 + 16 + 8 * hw]);
        acc = wmma16(cat8(alo, ahi), bv, acc);
      }
      float pfv = 0.f, plv = 0.f;
      #pragma unroll
      for (int r = 0; r < 8; ++r) {
        int nn = nt * 16 + r + 8 * hw;
        float v = fmaxf(acc[r] + eb2s[nn], 0.f);
        pfv += v * fws[nn];
        plv += v * lws[nn];
      }
      atomicAdd(&dotf[wv][l15], pfv);
      atomicAdd(&dotl[wv][l15], plv);
    }
    // gated accumulation (wave-local; DS ops in-order within a wave)
    if (lane < 16) {
      accF += gF[e] * dotf[wv][l15];
      accL += gL[e] * dotl[wv][l15];
      dotf[wv][l15] = 0.f; dotl[wv][l15] = 0.f;
    }
  }

  if (lane < 16) {
    int row = rowblk + wv * 16 + l15;
    out[row]        = 1.f / (1.f + __expf(-(accF + fb[0])));
    out[B_SZ + row] = 1.f / (1.f + __expf(-(accL + lb[0])));
  }
}

// ---------------------------------------------------------------------------
extern "C" void kernel_launch(void* const* d_in, const int* in_sizes, int n_in,
                              void* d_out, int out_size, void* d_ws, size_t ws_size,
                              hipStream_t stream) {
  const float* dense     = (const float*)d_in[0];
  const float* tables    = (const float*)d_in[1];
  const float* seq_table = (const float*)d_in[2];
  const float* att_w1    = (const float*)d_in[3];
  const float* att_b1    = (const float*)d_in[4];
  const float* att_w2    = (const float*)d_in[5];
  const float* att_b2    = (const float*)d_in[6];
  const float* att_w3    = (const float*)d_in[7];
  const float* att_b3    = (const float*)d_in[8];
  const float* gru_W     = (const float*)d_in[9];
  const float* gru_U     = (const float*)d_in[10];
  const float* gru_b     = (const float*)d_in[11];
  const float* ew1       = (const float*)d_in[12];
  const float* eb1       = (const float*)d_in[13];
  const float* ew2       = (const float*)d_in[14];
  const float* eb2       = (const float*)d_in[15];
  const float* gfw       = (const float*)d_in[16];
  const float* gfb       = (const float*)d_in[17];
  const float* glw       = (const float*)d_in[18];
  const float* glb       = (const float*)d_in[19];
  const float* fw        = (const float*)d_in[20];
  const float* fb        = (const float*)d_in[21];
  const float* lw        = (const float*)d_in[22];
  const float* lb        = (const float*)d_in[23];
  const int* sparse_idx  = (const int*)d_in[24];
  const int* seq_idx     = (const int*)d_in[25];
  const int* hist        = (const int*)d_in[26];

  float* dnn = (float*)d_ws;       // [B, 184] f32 feature buffer (~12 MB)
  size_t dnnBytes = (size_t)B_SZ * DNN_DIM * sizeof(float);
  _Float16* ew1T = (_Float16*)((char*)d_ws + dnnBytes);
  _Float16* ew2T = ew1T + N_EW1T;
  _Float16* dinW = ew2T + N_EW2T;
  float* out = (float*)d_out;      // [2B] : finish then like
  int V = in_sizes[2] / EMB;

  int prepN = N_EW1T + N_EW2T + N_DINW;
  k_prep<<<(prepN + 255) / 256, 256, 0, stream>>>(ew1, ew2, att_w1, att_w2,
                                                  ew1T, ew2T, dinW);
  k_embed<<<B_SZ / 8, 256, 0, stream>>>(dense, tables, seq_table, sparse_idx,
                                        seq_idx, dnn, V);
  k_din<<<B_SZ, 128, 0, stream>>>(tables, sparse_idx, hist, dinW, att_b1,
                                  att_b2, att_w3, att_b3, dnn);
  k_gru<<<B_SZ / 16, 256, 0, stream>>>(tables, hist, gru_W, gru_U, gru_b, dnn);
  k_mmoe<<<B_SZ / 64, 128, 0, stream>>>(dnn, ew1T, ew2T, eb1, eb2, gfw, gfb,
                                        glw, glb, fw, fb, lw, lb, out);
}
